// TemporalGNN_11467562680922
// MI455X (gfx1250) — matrix operand order repaired and checked by reading, not compiled, not run on hardware
//
#include <hip/hip_runtime.h>
#include <hip/hip_bf16.h>
#include <stdint.h>

// ---------------------------------------------------------------------------
// Problem constants (from reference)
// ---------------------------------------------------------------------------
#define TT   12
#define NN   10000
#define DD   256
#define HH   512
#define EE   160000
#define H3   1536          // 3*H
#define NP   10048         // N padded to multiple of 64
#define NCONV 2

typedef __bf16 bf16_t;
typedef __attribute__((ext_vector_type(16))) __bf16 v16bf;
typedef __attribute__((ext_vector_type(8)))  __bf16 v8bf;
typedef __attribute__((ext_vector_type(8)))  float  v8f;

// float -> bf16 (round to nearest even)
static __device__ __forceinline__ bf16_t f2bf(float f) {
    union { float f; unsigned u; } a; a.f = f;
    unsigned u = a.u + 0x7FFFu + ((a.u >> 16) & 1u);
    union { unsigned short s; bf16_t b; } r;
    r.s = (unsigned short)(u >> 16);
    return r.b;
}

// ---------------------------------------------------------------------------
// Utility kernels
// ---------------------------------------------------------------------------
__global__ void k_zero_u32(unsigned* __restrict__ p, long n) {
    long i = (long)blockIdx.x * blockDim.x + threadIdx.x;
    if (i < n) p[i] = 0u;
}

__global__ void k_cvt_bf16(const float* __restrict__ s, bf16_t* __restrict__ d, long n) {
    long i = (long)blockIdx.x * blockDim.x + threadIdx.x;
    if (i < n) d[i] = f2bf(s[i]);
}

// conv_W[l] is (H,H) used as h @ W; our GEMM computes A @ B^T with B (M,K)
// row-major, so store B'[j,k] = W[k,j] (transpose) per layer.
__global__ void k_cvt_transpose(const float* __restrict__ s, bf16_t* __restrict__ d) {
    long i = (long)blockIdx.x * blockDim.x + threadIdx.x;
    long tot = (long)NCONV * HH * HH;
    if (i >= tot) return;
    int l = (int)(i / (HH * HH));
    int r = (int)(i % (HH * HH));
    int k = r / HH, j = r % HH;
    d[(long)l * HH * HH + (long)j * HH + k] = f2bf(s[(long)l * HH * HH + (long)k * HH + j]);
}

// ---------------------------------------------------------------------------
// Graph normalization precompute
// ---------------------------------------------------------------------------
__global__ void k_deg_accum(const long long* __restrict__ ei, const float* __restrict__ w,
                            float* __restrict__ deg) {
    int e = blockIdx.x * blockDim.x + threadIdx.x;
    if (e >= EE) return;
    long long c = ei[(long long)EE + e];
    atomicAdd(&deg[c], w[e]);
}

__global__ void k_deg_finalize(float* __restrict__ deg, float* __restrict__ dinv,
                               float* __restrict__ selfw) {
    int n = blockIdx.x * blockDim.x + threadIdx.x;
    if (n >= NN) return;
    float dv = rsqrtf(deg[n] + 1.0f);
    dinv[n]  = dv;
    selfw[n] = dv * dv;
}

__global__ void k_edge_norm(const long long* __restrict__ ei, const float* __restrict__ w,
                            const float* __restrict__ dinv, float* __restrict__ norm) {
    int e = blockIdx.x * blockDim.x + threadIdx.x;
    if (e >= EE) return;
    long long r = ei[e];
    long long c = ei[(long long)EE + e];
    norm[e] = dinv[r] * w[e] * dinv[c];
}

// ---------------------------------------------------------------------------
// WMMA bf16 GEMM:  C(NP x M) = A(NP x K) @ B(M x K)^T (+ bias[M])
// block = 128 threads = 4 waves; block tile 64x64; wave tile 32x32 = 2x2 WMMA
// ---------------------------------------------------------------------------
__global__ __launch_bounds__(128)
void k_gemm_bf16(const bf16_t* __restrict__ A, const bf16_t* __restrict__ B,
                 const float* __restrict__ bias, float* __restrict__ C,
                 int K, int M) {
    const int lane = threadIdx.x & 31;
    const int wave = threadIdx.x >> 5;
    const int rowBase = blockIdx.y * 64 + (wave >> 1) * 32;   // node rows
    const int colBase = blockIdx.x * 64 + (wave & 1) * 32;    // output feats
    const int m_l  = lane & 15;
    const int kgrp = lane >> 4;          // 0 or 1

    v8f acc[2][2];
    #pragma unroll
    for (int i = 0; i < 2; ++i)
        #pragma unroll
        for (int j = 0; j < 2; ++j)
            acc[i][j] = (v8f){0.f,0.f,0.f,0.f,0.f,0.f,0.f,0.f};

    // A lane base: row = rowBase + i*16 + m_l ; holds K = kgrp*16+{0..15}
    const bf16_t* a0 = A + (size_t)(rowBase + m_l) * K + kgrp * 16;
    const bf16_t* a1 = a0 + (size_t)16 * K;
    // B lane base: col = colBase + j*16 + m_l ; holds K = kgrp*16+{0..15}
    const bf16_t* b0 = B + (size_t)(colBase + m_l) * K + kgrp * 16;
    const bf16_t* b1 = b0 + (size_t)16 * K;

    for (int k0 = 0; k0 < K; k0 += 32) {
        v16bf af[2], bfg[2];
        {
            v8bf lo = *(const v8bf*)(a0 + k0);
            v8bf hi = *(const v8bf*)(a0 + k0 + 8);
            #pragma unroll
            for (int i = 0; i < 8; ++i) { af[0][i] = lo[i]; af[0][i + 8] = hi[i]; }
        }
        {
            v8bf lo = *(const v8bf*)(a1 + k0);
            v8bf hi = *(const v8bf*)(a1 + k0 + 8);
            #pragma unroll
            for (int i = 0; i < 8; ++i) { af[1][i] = lo[i]; af[1][i + 8] = hi[i]; }
        }
        bfg[0] = *(const v16bf*)(b0 + k0);
        bfg[1] = *(const v16bf*)(b1 + k0);

        #pragma unroll
        for (int i = 0; i < 2; ++i)
            #pragma unroll
            for (int j = 0; j < 2; ++j)
                acc[i][j] = __builtin_amdgcn_wmma_f32_16x16x32_bf16(
                    false, af[i], false, bfg[j], (short)0, acc[i][j], false, false);
    }

    // C/D layout: VGPR r, lanes 0-15 -> (M=r, N=lane); lanes 16-31 -> (M=8+r)
    #pragma unroll
    for (int i = 0; i < 2; ++i) {
        #pragma unroll
        for (int j = 0; j < 2; ++j) {
            int n = colBase + j * 16 + m_l;
            float bv = bias ? bias[n] : 0.0f;
            int m0 = rowBase + i * 16 + kgrp * 8;
            float* cp = C + (size_t)m0 * M + n;
            #pragma unroll
            for (int r = 0; r < 8; ++r)
                cp[(size_t)r * M] = acc[i][j][r] + bv;
        }
    }
}

// ---------------------------------------------------------------------------
// GRU gate fusion: h = (1-z)*n + z*h ; also emits bf16 copy of h
// ---------------------------------------------------------------------------
__global__ void k_gru_gate(const float* __restrict__ gi, const float* __restrict__ gh,
                           float* __restrict__ h, bf16_t* __restrict__ hbf) {
    long idx = (long)blockIdx.x * blockDim.x + threadIdx.x;
    if (idx >= (long)NN * HH) return;
    int n = (int)(idx >> 9);      // /512
    int f = (int)(idx & 511);
    long gb = (long)n * H3 + f;
    float i_r = gi[gb], i_z = gi[gb + HH], i_n = gi[gb + 2 * HH];
    float h_r = gh[gb], h_z = gh[gb + HH], h_n = gh[gb + 2 * HH];
    float r = 1.0f / (1.0f + __expf(-(i_r + h_r)));
    float z = 1.0f / (1.0f + __expf(-(i_z + h_z)));
    float nn = tanhf(i_n + r * h_n);
    float hp = h[idx];
    float hv = (1.0f - z) * nn + z * hp;
    h[idx]  = hv;
    hbf[idx] = f2bf(hv);
}

// ---------------------------------------------------------------------------
// Edge message scatter: agg[col] += norm * hW[row]   (one block per edge)
// ---------------------------------------------------------------------------
__global__ __launch_bounds__(128)
void k_edge_msg(const long long* __restrict__ ei, const float* __restrict__ norm,
                const float* __restrict__ hW, float* __restrict__ agg) {
    int e = blockIdx.x;
    int f = threadIdx.x * 4;
    long long r = ei[e];
    long long c = ei[(long long)EE + e];
    float nm = norm[e];
    const float4 v = *(const float4*)(hW + (size_t)r * HH + f);
    float* ap = agg + (size_t)c * HH + f;
    atomicAdd(ap + 0, nm * v.x);
    atomicAdd(ap + 1, nm * v.y);
    atomicAdd(ap + 2, nm * v.z);
    atomicAdd(ap + 3, nm * v.w);
}

// ---------------------------------------------------------------------------
// GCN combine: h = relu(agg + selfw[n]*hW + b[f]) ; emits bf16 copy
// ---------------------------------------------------------------------------
__global__ void k_gcn_combine(const float* __restrict__ agg, const float* __restrict__ hW,
                              const float* __restrict__ selfw, const float* __restrict__ bias,
                              float* __restrict__ h, bf16_t* __restrict__ hbf) {
    long idx = (long)blockIdx.x * blockDim.x + threadIdx.x;
    if (idx >= (long)NN * HH) return;
    int n = (int)(idx >> 9);
    int f = (int)(idx & 511);
    float v = agg[idx] + selfw[n] * hW[idx] + bias[f];
    v = fmaxf(v, 0.0f);
    h[idx]  = v;
    hbf[idx] = f2bf(v);
}

// ---------------------------------------------------------------------------
// Final projection: out[n] = dot(h[n,:], lin_W) + lin_b   (one wave per node)
// ---------------------------------------------------------------------------
__global__ __launch_bounds__(128)
void k_out_dot(const float* __restrict__ h, const float* __restrict__ lw,
               const float* __restrict__ lb, float* __restrict__ out) {
    int node = blockIdx.x * 4 + (threadIdx.x >> 5);
    int lane = threadIdx.x & 31;
    if (node >= NN) return;
    const float* hp = h + (size_t)node * HH;
    float s = 0.0f;
    #pragma unroll
    for (int i = 0; i < 16; ++i)
        s += hp[i * 32 + lane] * lw[i * 32 + lane];
    #pragma unroll
    for (int o = 16; o > 0; o >>= 1)
        s += __shfl_down(s, o, 32);
    if (lane == 0) out[node] = s + lb[0];
}

// ---------------------------------------------------------------------------
// Host orchestration
// ---------------------------------------------------------------------------
extern "C" void kernel_launch(void* const* d_in, const int* in_sizes, int n_in,
                              void* d_out, int out_size, void* d_ws, size_t ws_size,
                              hipStream_t stream) {
    const float*     xs    = (const float*)d_in[0];      // (T,N,D)
    const long long* ei    = (const long long*)d_in[1];  // (2,E) int64
    const float*     ew    = (const float*)d_in[2];      // (E,)
    const float*     W_ih  = (const float*)d_in[3];      // (3H,D)
    const float*     W_hh  = (const float*)d_in[4];      // (3H,H)
    const float*     b_ih  = (const float*)d_in[5];      // (3H,)
    const float*     b_hh  = (const float*)d_in[6];      // (3H,)
    const float*     convW = (const float*)d_in[7];      // (2,H,H)
    const float*     convB = (const float*)d_in[8];      // (2,H)
    const float*     linW  = (const float*)d_in[9];      // (H,1)
    const float*     linB  = (const float*)d_in[10];     // (1,)
    float* out = (float*)d_out;                          // (T,N,1)

    // workspace carve-out (256B aligned)
    char* ws = (char*)d_ws;
    size_t off = 0;
    auto carve = [&](size_t bytes) -> void* {
        off = (off + 255) & ~(size_t)255;
        void* p = ws + off; off += bytes; return p;
    };
    float*  h_f32   = (float*)  carve((size_t)NP * HH * 4);
    bf16_t* h_bf    = (bf16_t*) carve((size_t)NP * HH * 2);
    bf16_t* x_bf    = (bf16_t*) carve((size_t)NP * DD * 2);
    float*  gi      = (float*)  carve((size_t)NP * H3 * 4);
    float*  gh      = (float*)  carve((size_t)NP * H3 * 4);
    float*  hW      = (float*)  carve((size_t)NP * HH * 4);
    float*  agg     = (float*)  carve((size_t)NP * HH * 4);
    bf16_t* Wih_bf  = (bf16_t*) carve((size_t)H3 * DD * 2);
    bf16_t* Whh_bf  = (bf16_t*) carve((size_t)H3 * HH * 2);
    bf16_t* Wcv_bf  = (bf16_t*) carve((size_t)NCONV * HH * HH * 2);
    float*  deg     = (float*)  carve((size_t)NN * 4);
    float*  dinv    = (float*)  carve((size_t)NN * 4);
    float*  selfw   = (float*)  carve((size_t)NN * 4);
    float*  normv   = (float*)  carve((size_t)EE * 4);

    const int B = 256;
    auto blocks = [](long n, int b) { return (unsigned)((n + b - 1) / b); };

    // --- one-time preprocessing (re-done every call: deterministic) ---
    k_zero_u32<<<blocks((long)NP * HH, B), B, 0, stream>>>((unsigned*)h_f32, (long)NP * HH);
    k_zero_u32<<<blocks((long)NP * HH / 2, B), B, 0, stream>>>((unsigned*)h_bf, (long)NP * HH / 2);
    k_zero_u32<<<blocks((long)NP * DD / 2, B), B, 0, stream>>>((unsigned*)x_bf, (long)NP * DD / 2);
    k_zero_u32<<<blocks(NN, B), B, 0, stream>>>((unsigned*)deg, NN);

    k_cvt_bf16<<<blocks((long)H3 * DD, B), B, 0, stream>>>(W_ih, Wih_bf, (long)H3 * DD);
    k_cvt_bf16<<<blocks((long)H3 * HH, B), B, 0, stream>>>(W_hh, Whh_bf, (long)H3 * HH);
    k_cvt_transpose<<<blocks((long)NCONV * HH * HH, B), B, 0, stream>>>(convW, Wcv_bf);

    k_deg_accum<<<blocks(EE, B), B, 0, stream>>>(ei, ew, deg);
    k_deg_finalize<<<blocks(NN, B), B, 0, stream>>>(deg, dinv, selfw);
    k_edge_norm<<<blocks(EE, B), B, 0, stream>>>(ei, ew, dinv, normv);

    dim3 gemmBlk(128);
    dim3 gGI(H3 / 64, NP / 64);   // (24,157)
    dim3 gCV(HH / 64, NP / 64);   // (8,157)

    for (int t = 0; t < TT; ++t) {
        // x_t -> bf16 (pad rows already zero)
        k_cvt_bf16<<<blocks((long)NN * DD, B), B, 0, stream>>>(
            xs + (size_t)t * NN * DD, x_bf, (long)NN * DD);

        // GRU GEMMs: gi = x @ W_ih^T + b_ih ; gh = h @ W_hh^T + b_hh
        k_gemm_bf16<<<gGI, gemmBlk, 0, stream>>>(x_bf, Wih_bf, b_ih, gi, DD, H3);
        k_gemm_bf16<<<gGI, gemmBlk, 0, stream>>>(h_bf, Whh_bf, b_hh, gh, HH, H3);
        k_gru_gate<<<blocks((long)NN * HH, B), B, 0, stream>>>(gi, gh, h_f32, h_bf);

        // 2 GCN layers
        for (int l = 0; l < NCONV; ++l) {
            k_gemm_bf16<<<gCV, gemmBlk, 0, stream>>>(
                h_bf, Wcv_bf + (size_t)l * HH * HH, nullptr, hW, HH, HH);
            k_zero_u32<<<blocks((long)NN * HH, B), B, 0, stream>>>((unsigned*)agg, (long)NN * HH);
            k_edge_msg<<<EE, 128, 0, stream>>>(ei, normv, hW, agg);
            k_gcn_combine<<<blocks((long)NN * HH, B), B, 0, stream>>>(
                agg, hW, selfw, convB + (size_t)l * HH, h_f32, h_bf);
        }

        // out_t = h @ lin_W + lin_b
        k_out_dot<<<(NN + 3) / 4, 128, 0, stream>>>(h_f32, linW, linB, out + (size_t)t * NN);
    }
}
